// MLP3_71494025609415
// MI455X (gfx1250) — matrix operand order repaired and checked by reading, not compiled, not run on hardware
//
#include <hip/hip_runtime.h>

typedef __attribute__((ext_vector_type(16))) __bf16 v16bf;
typedef __attribute__((ext_vector_type(8)))  __bf16 v8bf;
typedef __attribute__((ext_vector_type(8)))  float  v8f;

#define BATCH  2048
#define INST   64
#define VOCAB  128
#define ED     256
#define HID    1024
#define NOUT   128

#define MTILE  128      // batch rows per workgroup (8 waves x 16 rows)
#define HCHUNK 64       // hidden columns per chunk (16 chunks)
#define NCHUNK (HID / HCHUNK)

// LDS row strides (bf16 units); every row start stays 32B-aligned for v16bf loads.
#define XSTR   272      // 544B rows
#define W1STR  272
#define USTR   80       // 160B rows
#define HSTR   80

// LDS layout (element offsets into dynamic-shared bf16 array; base assumed 0)
#define X_OFF   0
#define W1_0E   (MTILE * XSTR)            // 34816
#define W1_1E   (W1_0E + HCHUNK * W1STR)  // 52224
#define UT_0E   (W1_1E + HCHUNK * W1STR)  // 69632
#define UT_1E   (UT_0E + NOUT * USTR)     // 79872
#define HB_E    (UT_1E + NOUT * USTR)     // 90112
#define LDS_ELEMS (HB_E + 8 * 16 * HSTR)  // 100352 bf16 = 200704 B

// Byte offsets for async-to-LDS destinations (must match the element offsets above)
#define W1_0B   (W1_0E * 2)   // 69632
#define W1_1B   (W1_1E * 2)   // 104448
#define UT_0B   (UT_0E * 2)   // 139264
#define UT_1B   (UT_1E * 2)   // 159744

#define WS_W1_BYTES ((size_t)INST * HID * ED * 2)    // 32 MiB: W1^T  [i][n(1024)][k(256)]
#define WS_UT_BYTES ((size_t)INST * NOUT * HID * 2)  // 16 MiB: U^T   [i][n(128)][k(1024)]
#define WS_NEEDED   (WS_W1_BYTES + WS_UT_BYTES)

static __device__ __forceinline__ __bf16 f2bf(float f) {
  union { float f; unsigned u; } c; c.f = f;
  unsigned r = c.u + 0x7FFFu + ((c.u >> 16) & 1u);
  unsigned short h = (unsigned short)(r >> 16);
  return __builtin_bit_cast(__bf16, h);
}

// ---------------- Pre-pass: transpose+convert fp32 [i][K][N] -> bf16 [i][N][K] ----
__global__ __launch_bounds__(256)
void transpose_cvt_bf16(const float* __restrict__ src, __bf16* __restrict__ dst,
                        int K, int N)
{
  __shared__ __bf16 tile[64][72];
  const int i  = blockIdx.x;
  const int kb = blockIdx.y * 64;
  const int nb = blockIdx.z * 64;
  const float* s = src + (size_t)i * K * N;
  __bf16*      d = dst + (size_t)i * N * K;
  const int t64 = threadIdx.x & 63;
  const int q   = threadIdx.x >> 6;          // 0..3
  for (int kk = q * 16; kk < q * 16 + 16; ++kk)          // coalesced f32 reads along n
    tile[kk][t64] = f2bf(s[(size_t)(kb + kk) * N + nb + t64]);
  __syncthreads();
  for (int nn = q * 16; nn < q * 16 + 16; ++nn)          // coalesced bf16 writes along k
    d[(size_t)(nb + nn) * K + kb + t64] = tile[t64][nn];
}

// ------------- Async DMA of one chunk's weights into an LDS double-buffer -------
static __device__ __forceinline__ void issue_async_chunk(
    unsigned long long w1base, unsigned long long utbase, int hc,
    unsigned w1lds, unsigned utlds, int tid)
{
  const int wv = tid >> 5, lane = tid & 31;
  // W1^T chunk: rows n = hc..hc+63 of [1024][256] bf16 (512B rows, fully linear)
  unsigned long long gw = w1base + (unsigned long long)hc * 512ull
                          + (unsigned long long)tid * 16ull;
  unsigned lw = w1lds + (unsigned)wv * 544u + (unsigned)lane * 16u;
#pragma unroll
  for (int j = 0; j < 8; ++j) {                       // 8 rows per issue, 64 rows total
    asm volatile("global_load_async_to_lds_b128 %0, %1, off"
                 :: "v"(lw), "v"(gw) : "memory");
    gw += 4096ull;  lw += 8u * 544u;
  }
  // U^T chunk: 128B slice (k = hc..hc+63) of each of 128 rows (row stride 2048B)
  const int rn = tid >> 3, ro = (tid & 7) * 16;
  unsigned long long gu = utbase + (unsigned long long)hc * 2ull
                          + (unsigned long long)rn * 2048ull + (unsigned long long)ro;
  unsigned lu = utlds + (unsigned)rn * 160u + (unsigned)ro;
#pragma unroll
  for (int j = 0; j < 4; ++j) {                       // 32 rows per issue, 128 rows total
    asm volatile("global_load_async_to_lds_b128 %0, %1, off"
                 :: "v"(lu), "v"(gu) : "memory");
    gu += 32ull * 2048ull;  lu += 32u * 160u;
  }
}

// --------------------------------- Main fused kernel ----------------------------
__global__ __launch_bounds__(256)
void mlp3_fused_wmma(const int* __restrict__ a,
                     const float* __restrict__ embL,
                     const float* __restrict__ embR,
                     const float* __restrict__ lin,
                     const float* __restrict__ unemb,
                     const __bf16* __restrict__ w1t,
                     const __bf16* __restrict__ ut,
                     float* __restrict__ out,
                     int useAsync)
{
  extern __shared__ __attribute__((aligned(128))) __bf16 smem[];
  __bf16* Xs = smem + X_OFF;
  __bf16* Hb = smem + HB_E;

  const int i    = blockIdx.x;
  const int b0   = blockIdx.y * MTILE;
  const int tid  = threadIdx.x;
  const int lane = tid & 31;
  const int wv   = tid >> 5;
  const int lr   = lane & 15;
  const int hi   = lane >> 4;

  // ---- Stage X = bf16(embL[i,a1] + embR[i,a2]) ----
  {
    int r = tid >> 1, half = tid & 1;
    int b  = b0 + r;
    int a1 = a[2 * b], a2 = a[2 * b + 1];
    const float4* pl = (const float4*)(embL + ((size_t)i * VOCAB + a1) * ED) + half * 32;
    const float4* pr = (const float4*)(embR + ((size_t)i * VOCAB + a2) * ED) + half * 32;
    __bf16* xr = Xs + r * XSTR + half * 128;
#pragma unroll 4
    for (int e = 0; e < 32; ++e) {
      float4 l = pl[e], rr = pr[e];
      xr[4 * e + 0] = f2bf(l.x + rr.x);
      xr[4 * e + 1] = f2bf(l.y + rr.y);
      xr[4 * e + 2] = f2bf(l.z + rr.z);
      xr[4 * e + 3] = f2bf(l.w + rr.w);
    }
  }

  v8f oacc[8];
#pragma unroll
  for (int t = 0; t < 8; ++t) oacc[t] = (v8f)0.0f;

  const float* linI = lin   + (size_t)i * ED  * HID;
  const float* uniI = unemb + (size_t)i * HID * NOUT;
  const unsigned long long w1base =
      (unsigned long long)(const char*)(w1t + (size_t)i * HID * ED);
  const unsigned long long utbase =
      (unsigned long long)(const char*)(ut + (size_t)i * NOUT * HID);

  if (useAsync)
    issue_async_chunk(w1base, utbase, 0, W1_0B, UT_0B, tid);

  for (int c = 0; c < NCHUNK; ++c) {
    const int hc = c * HCHUNK;
    int cur;
    if (useAsync) {
      asm volatile("s_wait_asynccnt 0x0" ::: "memory");  // chunk c resident in LDS
      __syncthreads();                                    // everyone done with buf[(c+1)&1]
      if (c + 1 < NCHUNK)
        issue_async_chunk(w1base, utbase, hc + HCHUNK,
                          ((c + 1) & 1) ? W1_1B : W1_0B,
                          ((c + 1) & 1) ? UT_1B : UT_0B, tid);
      cur = c & 1;
    } else {
      __syncthreads();
      { // W1^T staging: linear[i][k][hc+n] -> buf0[n][k]
        int n = tid & 63, ks = (tid >> 6) * 64;
        const float* src = linI + hc + n;
        __bf16* dstp = smem + W1_0E;
#pragma unroll 8
        for (int k = ks; k < ks + 64; ++k)
          dstp[n * W1STR + k] = f2bf(src[(size_t)k * HID]);
      }
      { // U^T staging: unemb[i][hc+k][n] -> buf0[n][k]
        int n = tid & 127, ks = (tid >> 7) * 32;
        const float* src = uniI + (size_t)hc * NOUT + n;
        __bf16* dstp = smem + UT_0E;
#pragma unroll 8
        for (int k = ks; k < ks + 32; ++k)
          dstp[n * USTR + k] = f2bf(src[(size_t)k * NOUT]);
      }
      if (hc + HCHUNK < HID) {
        __builtin_prefetch(linI + (hc + HCHUNK) + (tid & 63) + (size_t)((tid >> 6) * 64) * HID, 0, 1);
        __builtin_prefetch(uniI + (size_t)(hc + HCHUNK + (tid >> 7) * 32) * NOUT + (tid & 127), 0, 1);
      }
      __syncthreads();
      cur = 0;
    }
    // Select double-buffer halves with pointer arithmetic only (no pointer arrays:
    // LDS-pointer aggregate initializers are unlinkable on this toolchain).
    const __bf16* W1c = smem + (cur ? W1_1E : W1_0E);
    const __bf16* Utc = smem + (cur ? UT_1E : UT_0E);

    // ---- GEMM1: hidden[16x64] = X[16x256] @ W1[256x64] ----
    v8f hacc[4];
#pragma unroll
    for (int t = 0; t < 4; ++t) hacc[t] = (v8f)0.0f;

    const __bf16* xrow = Xs + (wv * 16 + lr) * XSTR + hi * 16;
#pragma unroll
    for (int ks = 0; ks < 8; ++ks) {
      const int k0 = ks * 32;
      v8bf alo = *(const v8bf*)(xrow + k0);
      v8bf ahi = *(const v8bf*)(xrow + k0 + 8);
      v16bf af = __builtin_shufflevector(alo, ahi,
                   0,1,2,3,4,5,6,7,8,9,10,11,12,13,14,15);
      v16bf bf0 = *(const v16bf*)(W1c + ( 0 + lr) * W1STR + k0 + hi * 16);
      v16bf bf1 = *(const v16bf*)(W1c + (16 + lr) * W1STR + k0 + hi * 16);
      v16bf bf2 = *(const v16bf*)(W1c + (32 + lr) * W1STR + k0 + hi * 16);
      v16bf bf3 = *(const v16bf*)(W1c + (48 + lr) * W1STR + k0 + hi * 16);
      hacc[0] = __builtin_amdgcn_wmma_f32_16x16x32_bf16(
                  false, af, false, bf0, (short)0, hacc[0], false, false);
      hacc[1] = __builtin_amdgcn_wmma_f32_16x16x32_bf16(
                  false, af, false, bf1, (short)0, hacc[1], false, false);
      hacc[2] = __builtin_amdgcn_wmma_f32_16x16x32_bf16(
                  false, af, false, bf2, (short)0, hacc[2], false, false);
      hacc[3] = __builtin_amdgcn_wmma_f32_16x16x32_bf16(
                  false, af, false, bf3, (short)0, hacc[3], false, false);
    }

    // ---- exact GELU, relayout D->A via per-wave LDS scratch ----
    __bf16* hb = Hb + wv * 16 * HSTR;
#pragma unroll
    for (int t = 0; t < 4; ++t) {
#pragma unroll
      for (int v = 0; v < 8; ++v) {
        float x = hacc[t][v];
        float g = 0.5f * x * (1.0f + erff(x * 0.70710678118654752f));
        hb[(v + hi * 8) * HSTR + t * 16 + lr] = f2bf(g);
      }
    }

    // ---- GEMM2: out[16x128] += gelu(hidden)[16x64] @ U[64x128] ----
    const __bf16* hrow = hb + lr * HSTR + hi * 16;
#pragma unroll
    for (int ks = 0; ks < 2; ++ks) {
      const int k0 = ks * 32;
      v8bf alo = *(const v8bf*)(hrow + k0);
      v8bf ahi = *(const v8bf*)(hrow + k0 + 8);
      v16bf af = __builtin_shufflevector(alo, ahi,
                   0,1,2,3,4,5,6,7,8,9,10,11,12,13,14,15);
      v16bf bf[8];
#pragma unroll
      for (int t = 0; t < 8; ++t)
        bf[t] = *(const v16bf*)(Utc + (t * 16 + lr) * USTR + k0 + hi * 16);
#pragma unroll
      for (int t = 0; t < 8; ++t)
        oacc[t] = __builtin_amdgcn_wmma_f32_16x16x32_bf16(
                    false, af, false, bf[t], (short)0, oacc[t], false, false);
    }
  }

  // ---- Epilogue ----
#pragma unroll
  for (int t = 0; t < 8; ++t) {
    int n = t * 16 + lr;
#pragma unroll
    for (int v = 0; v < 8; ++v) {
      int b = b0 + wv * 16 + v + hi * 8;
      out[((size_t)b * INST + i) * NOUT + n] = oacc[t][v];
    }
  }
}

extern "C" void kernel_launch(void* const* d_in, const int* in_sizes, int n_in,
                              void* d_out, int out_size, void* d_ws, size_t ws_size,
                              hipStream_t stream) {
  (void)in_sizes; (void)n_in; (void)out_size;
  const int*   a    = (const int*)d_in[0];
  const float* embL = (const float*)d_in[1];
  const float* embR = (const float*)d_in[2];
  const float* lin  = (const float*)d_in[3];
  const float* un   = (const float*)d_in[4];
  float* out = (float*)d_out;

  const int useAsync = (d_ws != nullptr && ws_size >= WS_NEEDED) ? 1 : 0;
  __bf16* w1t = (__bf16*)d_ws;
  __bf16* utp = (__bf16*)((char*)d_ws + WS_W1_BYTES);

  if (useAsync) {
    // One-time (per launch) fp32 -> bf16 transposed weight pre-pass into d_ws.
    transpose_cvt_bf16<<<dim3(INST, ED / 64, HID / 64), dim3(256), 0, stream>>>(
        lin, w1t, ED, HID);
    transpose_cvt_bf16<<<dim3(INST, HID / 64, NOUT / 64), dim3(256), 0, stream>>>(
        un, utp, HID, NOUT);
  }

  dim3 grid(INST, BATCH / MTILE, 1);       // 64 x 16 workgroups
  dim3 block(256, 1, 1);                   // 8 wave32
  size_t ldsBytes = (size_t)LDS_ELEMS * 2; // ~196 KB dynamic LDS
  mlp3_fused_wmma<<<grid, block, ldsBytes, stream>>>(
      a, embL, embR, lin, un, w1t, utp, out, useAsync);
}
